// DocREModel_11038065951139
// MI455X (gfx1250) — compile-verified, matrix operand order checked by reading, not run on hardware
//
#include <hip/hip_runtime.h>
#include <hip/hip_bf16.h>

// ---- Problem constants -----------------------------------------------------
#define B_   4
#define L_   512
#define H_   768
#define NH_  12
#define E_   32
#define M_   4
#define P_   992
#define NTOT (B_ * P_)      // 3968
#define EMB_ 768
#define BLK_ 64
#define KBIL (EMB_ * BLK_)  // 49152
#define NCLS 97
#define NCLSPAD 112         // 7 tiles of 16

// ---- WMMA types (gfx1250, wave32) -----------------------------------------
typedef __attribute__((ext_vector_type(16))) __bf16 v16bf;
typedef __attribute__((ext_vector_type(8)))  float  v8f;

union Frag16 { unsigned u[8]; v16bf v; };

__device__ __forceinline__ unsigned short f2bf(float f) {
  unsigned u = __float_as_uint(f);
  return (unsigned short)((u + 0x7FFFu + ((u >> 16) & 1u)) >> 16);
}
__device__ __forceinline__ float bf2f(unsigned short h) {
  return __uint_as_float(((unsigned)h) << 16);
}

// Load one 16x32 bf16 A (or B^T) fragment for this lane.
// K-half p sits at K = p + 8*((p>=8)+grp)  ->  dword p2 at 2*p2 + 8*grp + (p2>=4?8:0)
// Compiler vectorizes this into two global_load_b128 (bytes [0..15] and [32..47] +16*grp).
__device__ __forceinline__ v16bf load_frag32(const unsigned short* row, int kk, int grp) {
  Frag16 f;
  const unsigned short* p = row + kk + 8 * grp;
#pragma unroll
  for (int p2 = 0; p2 < 8; ++p2) {
    int K0 = 2 * p2 + (p2 >= 4 ? 8 : 0);
    f.u[p2] = *(const unsigned*)(p + K0);
  }
  return f.v;
}

__device__ __forceinline__ v8f wmma_bf16(v16bf a, v16bf b, v8f c) {
  return __builtin_amdgcn_wmma_f32_16x16x32_bf16(false, a, false, b, (short)0, c, false, false);
}

// 2x2 register-blocked tile product: C[2][2] (each 16x16) over K.
// A: M-major rows of K-contiguous bf16; BT: N-major rows of K-contiguous bf16.
__device__ __forceinline__ void gemm_tile2x2(const unsigned short* A, int lda,
                                             const unsigned short* BT, int ldb,
                                             int rowBase, int colBase, int K,
                                             v8f acc[2][2]) {
  int lane = threadIdx.x & 31;
  int rc   = lane & 15;
  int grp  = (lane >> 4) & 1;
  const unsigned short* a0 = A  + (size_t)(rowBase + rc) * (size_t)lda;
  const unsigned short* a1 = a0 + (size_t)16 * (size_t)lda;
  const unsigned short* b0 = BT + (size_t)(colBase + rc) * (size_t)ldb;
  const unsigned short* b1 = b0 + (size_t)16 * (size_t)ldb;
  for (int kk = 0; kk < K; kk += 32) {
    v16bf av0 = load_frag32(a0, kk, grp);
    v16bf av1 = load_frag32(a1, kk, grp);
    v16bf bv0 = load_frag32(b0, kk, grp);
    v16bf bv1 = load_frag32(b1, kk, grp);
    acc[0][0] = wmma_bf16(av0, bv0, acc[0][0]);
    acc[0][1] = wmma_bf16(av0, bv1, acc[0][1]);
    acc[1][0] = wmma_bf16(av1, bv0, acc[1][0]);
    acc[1][1] = wmma_bf16(av1, bv1, acc[1][1]);
  }
}

// ---- K0: fp32 -> bf16 transpose (+ zero pad columns) -----------------------
__global__ void transpose_bf16_kernel(const float* __restrict__ src,
                                      unsigned short* __restrict__ dst,
                                      int R, int C, int Cpad,
                                      long srcStride, long dstStride) {
  long batch = blockIdx.z;
  long idx = (long)blockIdx.x * blockDim.x + threadIdx.x;
  long total = (long)R * (long)Cpad;
  if (idx >= total) return;
  int r = (int)(idx % R);
  int c = (int)(idx / R);
  float v = (c < C) ? src[batch * srcStride + (long)r * C + c] : 0.0f;
  dst[batch * dstStride + idx] = f2bf(v);
}

// ---- K1: ent_emb (logsumexp over M) and ent_att (mean over M) --------------
__global__ __launch_bounds__(256)
void ent_kernel(const float* __restrict__ seq, const float* __restrict__ att,
                const int* __restrict__ mpos,
                float* __restrict__ ent_emb, float* __restrict__ ent_att) {
  int be = blockIdx.x;
  int b = be / E_, e = be % E_;
  int mi[M_];
#pragma unroll
  for (int m = 0; m < M_; ++m) mi[m] = mpos[(b * E_ + e) * M_ + m];

  const float* sb = seq + (size_t)b * L_ * H_;
  for (int h = threadIdx.x; h < H_; h += blockDim.x) {
    float x0 = sb[(size_t)mi[0] * H_ + h];
    float x1 = sb[(size_t)mi[1] * H_ + h];
    float x2 = sb[(size_t)mi[2] * H_ + h];
    float x3 = sb[(size_t)mi[3] * H_ + h];
    float mx = fmaxf(fmaxf(x0, x1), fmaxf(x2, x3));
    float s = __expf(x0 - mx) + __expf(x1 - mx) + __expf(x2 - mx) + __expf(x3 - mx);
    ent_emb[((size_t)b * E_ + e) * H_ + h] = mx + __logf(s);
  }

  const float* ab = att + (size_t)b * NH_ * L_ * L_;
  for (int t = threadIdx.x; t < NH_ * L_; t += blockDim.x) {
    int nh = t / L_, l = t % L_;
    float s = 0.0f;
#pragma unroll
    for (int m = 0; m < M_; ++m)
      s += ab[((size_t)nh * L_ + mi[m]) * L_ + l];
    ent_att[(((size_t)b * E_ + e) * NH_ + nh) * L_ + l] = s * 0.25f;
  }
}

// ---- K2: ht_att = normalize(mean_nh(h_att * t_att)) -> bf16 ---------------
__global__ __launch_bounds__(256)
void htatt_kernel(const int* __restrict__ hts, const float* __restrict__ ent_att,
                  unsigned short* __restrict__ htatt_bf) {
  int bp = blockIdx.x;
  int b = bp / P_, p = bp % P_;
  int h0 = hts[((size_t)b * P_ + p) * 2 + 0];
  int t0 = hts[((size_t)b * P_ + p) * 2 + 1];
  const float* ha = ent_att + ((size_t)b * E_ + h0) * NH_ * L_;
  const float* ta = ent_att + ((size_t)b * E_ + t0) * NH_ * L_;

  __shared__ float red[256];
  __shared__ float wbuf[L_];
  float lsum = 0.0f;
  for (int l = threadIdx.x; l < L_; l += 256) {
    float s = 0.0f;
#pragma unroll
    for (int nh = 0; nh < NH_; ++nh) s += ha[nh * L_ + l] * ta[nh * L_ + l];
    s *= (1.0f / (float)NH_);
    wbuf[l] = s;
    lsum += s;
  }
  red[threadIdx.x] = lsum;
  __syncthreads();
  for (int off = 128; off > 0; off >>= 1) {
    if (threadIdx.x < (unsigned)off) red[threadIdx.x] += red[threadIdx.x + off];
    __syncthreads();
  }
  float inv = 1.0f / (red[0] + 1e-5f);
  for (int l = threadIdx.x; l < L_; l += 256)
    htatt_bf[((size_t)b * P_ + p) * L_ + l] = f2bf(wbuf[l] * inv);
}

// ---- K3: rs = ht_att @ seq (per batch, WMMA bf16, 2x2 blocked) ------------
__global__ __launch_bounds__(32)
void gemm_rs_kernel(const unsigned short* __restrict__ htatt,
                    const unsigned short* __restrict__ seqT,
                    float* __restrict__ rs) {
  int b = blockIdx.z;
  const unsigned short* A  = htatt + (size_t)b * P_ * L_;   // P x 512
  const unsigned short* BT = seqT  + (size_t)b * H_ * L_;   // 768 rows of 512
  v8f acc[2][2] = {};
  gemm_tile2x2(A, L_, BT, L_, blockIdx.x * 32, blockIdx.y * 32, L_, acc);
  int lane = threadIdx.x & 31;
  int grp  = (lane >> 4) & 1;
  float* C = rs + (size_t)b * P_ * H_;
#pragma unroll
  for (int tm = 0; tm < 2; ++tm)
#pragma unroll
    for (int tn = 0; tn < 2; ++tn) {
      int col = blockIdx.y * 32 + tn * 16 + (lane & 15);
      int row0 = blockIdx.x * 32 + tm * 16 + 8 * grp;
#pragma unroll
      for (int v = 0; v < 8; ++v)
        C[(size_t)(row0 + v) * H_ + col] = acc[tm][tn][v];
    }
}

// ---- K4: build concat inputs Ah=[hs|rs], At=[ts|rs] in bf16 ---------------
__global__ __launch_bounds__(256)
void build_A_kernel(const int* __restrict__ hts, const float* __restrict__ ent_emb,
                    const float* __restrict__ rs,
                    unsigned short* __restrict__ Ah, unsigned short* __restrict__ At) {
  int n = blockIdx.x;
  int b = n / P_, p = n % P_;
  int h0 = hts[((size_t)b * P_ + p) * 2 + 0];
  int t0 = hts[((size_t)b * P_ + p) * 2 + 1];
  const float* he = ent_emb + ((size_t)b * E_ + h0) * H_;
  const float* te = ent_emb + ((size_t)b * E_ + t0) * H_;
  const float* r  = rs + (size_t)n * H_;
  for (int c = threadIdx.x; c < H_; c += 256) {
    unsigned short rb = f2bf(r[c]);
    Ah[(size_t)n * 1536 + c]       = f2bf(he[c]);
    Ah[(size_t)n * 1536 + 768 + c] = rb;
    At[(size_t)n * 1536 + c]       = f2bf(te[c]);
    At[(size_t)n * 1536 + 768 + c] = rb;
  }
}

// ---- K5: hs_proj / ts_proj = tanh(A @ W + b), bf16 out (WMMA, 2x2) --------
__global__ __launch_bounds__(32)
void gemm_proj_kernel(const unsigned short* __restrict__ Ah,
                      const unsigned short* __restrict__ At,
                      const unsigned short* __restrict__ WhT,
                      const unsigned short* __restrict__ WtT,
                      const float* __restrict__ bh, const float* __restrict__ bt,
                      unsigned short* __restrict__ hsp, unsigned short* __restrict__ tsp) {
  int z = blockIdx.z;
  const unsigned short* A  = z ? At  : Ah;
  const unsigned short* BT = z ? WtT : WhT;
  const float* bias        = z ? bt  : bh;
  unsigned short* out      = z ? tsp : hsp;
  v8f acc[2][2] = {};
  gemm_tile2x2(A, 1536, BT, 1536, blockIdx.x * 32, blockIdx.y * 32, 1536, acc);
  int lane = threadIdx.x & 31;
  int grp  = (lane >> 4) & 1;
#pragma unroll
  for (int tm = 0; tm < 2; ++tm)
#pragma unroll
    for (int tn = 0; tn < 2; ++tn) {
      int col = blockIdx.y * 32 + tn * 16 + (lane & 15);
      int row0 = blockIdx.x * 32 + tm * 16 + 8 * grp;
      float bv = bias[col];
#pragma unroll
      for (int v = 0; v < 8; ++v)
        out[(size_t)(row0 + v) * EMB_ + col] = f2bf(tanhf(acc[tm][tn][v] + bv));
    }
}

// ---- K6: logits = fused bilinear @ Wb + bb --------------------------------
// A row n, column k*4096+i*64+j = hs_proj[n, k*64+i] * ts_proj[n, k*64+j],
// built per-fragment from a 32-row LDS stage.  7 waves = 7 column tiles;
// each wave covers 2 row tiles so every Wb fragment feeds 2 WMMAs.
__global__ __launch_bounds__(224)
void gemm_logits_kernel(const unsigned short* __restrict__ hsp,
                        const unsigned short* __restrict__ tsp,
                        const unsigned short* __restrict__ WbT,
                        const float* __restrict__ bb, float* __restrict__ out) {
  __shared__ unsigned short hsr[32 * EMB_];   // 48 KB
  __shared__ unsigned short tsr[32 * EMB_];   // 48 KB
  int blockRow = blockIdx.x;                  // 32 rows of N

  {
    const unsigned* gh = (const unsigned*)(hsp + (size_t)blockRow * 32 * EMB_);
    const unsigned* gt = (const unsigned*)(tsp + (size_t)blockRow * 32 * EMB_);
    unsigned* sh = (unsigned*)hsr;
    unsigned* st = (unsigned*)tsr;
    for (int i = threadIdx.x; i < 32 * EMB_ / 2; i += blockDim.x) {
      sh[i] = gh[i];
      st[i] = gt[i];
    }
  }
  __syncthreads();

  int wave = threadIdx.x >> 5;   // column tile 0..6
  int lane = threadIdx.x & 31;
  int r    = lane & 15;
  int grp  = (lane >> 4) & 1;
  const unsigned short* brow   = WbT + (size_t)(wave * 16 + r) * KBIL;
  const unsigned short* hsrow0 = hsr + r * EMB_;
  const unsigned short* hsrow1 = hsr + (16 + r) * EMB_;
  const unsigned short* tsrow0 = tsr + r * EMB_;
  const unsigned short* tsrow1 = tsr + (16 + r) * EMB_;

  v8f acc0 = {}, acc1 = {};
  for (int kk = 0; kk < KBIL; kk += 32) {
    int kb  = kk >> 12;        // bilinear block 0..11
    int rem = kk & 4095;
    int i   = rem >> 6;        // 0..63
    int j0  = rem & 63;        // 0 or 32
    int jbase = kb * BLK_ + j0 + 8 * grp;
    float s0 = bf2f(hsrow0[kb * BLK_ + i]);
    float s1 = bf2f(hsrow1[kb * BLK_ + i]);
    const unsigned* tp0 = (const unsigned*)(tsrow0 + jbase);
    const unsigned* tp1 = (const unsigned*)(tsrow1 + jbase);
    Frag16 af0, af1;
#pragma unroll
    for (int p2 = 0; p2 < 8; ++p2) {
      int K0 = 2 * p2 + (p2 >= 4 ? 8 : 0);
      unsigned d0 = tp0[K0 >> 1];
      unsigned d1 = tp1[K0 >> 1];
      unsigned short l0 = f2bf(s0 * bf2f((unsigned short)(d0 & 0xFFFFu)));
      unsigned short h0v = f2bf(s0 * bf2f((unsigned short)(d0 >> 16)));
      unsigned short l1 = f2bf(s1 * bf2f((unsigned short)(d1 & 0xFFFFu)));
      unsigned short h1v = f2bf(s1 * bf2f((unsigned short)(d1 >> 16)));
      af0.u[p2] = ((unsigned)h0v << 16) | (unsigned)l0;
      af1.u[p2] = ((unsigned)h1v << 16) | (unsigned)l1;
    }
    v16bf bv = load_frag32(brow, kk, grp);
    acc0 = wmma_bf16(af0.v, bv, acc0);
    acc1 = wmma_bf16(af1.v, bv, acc1);
  }

  int c = wave * 16 + (lane & 15);
  if (c < NCLS) {
    float bbv = bb[c];
    int n0 = blockRow * 32 + 8 * grp;
#pragma unroll
    for (int v = 0; v < 8; ++v) {
      out[(size_t)(n0 + v) * NCLS + c]      = acc0[v] + bbv;
      out[(size_t)(n0 + 16 + v) * NCLS + c] = acc1[v] + bbv;
    }
  }
}

// ---- Host-side orchestration ----------------------------------------------
extern "C" void kernel_launch(void* const* d_in, const int* in_sizes, int n_in,
                              void* d_out, int out_size, void* d_ws, size_t ws_size,
                              hipStream_t stream) {
  (void)in_sizes; (void)n_in; (void)out_size; (void)ws_size;
  const float* seq  = (const float*)d_in[0];
  const float* att  = (const float*)d_in[1];
  const int*   mpos = (const int*)d_in[2];
  const int*   hts  = (const int*)d_in[3];
  const float* Wh   = (const float*)d_in[4];
  const float* bh   = (const float*)d_in[5];
  const float* Wt   = (const float*)d_in[6];
  const float* bt   = (const float*)d_in[7];
  const float* Wb   = (const float*)d_in[8];
  const float* bb   = (const float*)d_in[9];
  float* out = (float*)d_out;

  char* ws = (char*)d_ws;
  size_t off = 0;
  float* ent_emb = (float*)(ws + off);            off += (size_t)B_ * E_ * H_ * 4;
  float* ent_att = (float*)(ws + off);            off += (size_t)B_ * E_ * NH_ * L_ * 4;
  unsigned short* htatt = (unsigned short*)(ws + off); off += (size_t)B_ * P_ * L_ * 2;
  unsigned short* seqT  = (unsigned short*)(ws + off); off += (size_t)B_ * H_ * L_ * 2;
  unsigned short* WhT   = (unsigned short*)(ws + off); off += (size_t)H_ * 1536 * 2;
  unsigned short* WtT   = (unsigned short*)(ws + off); off += (size_t)H_ * 1536 * 2;
  unsigned short* WbT   = (unsigned short*)(ws + off); off += (size_t)NCLSPAD * KBIL * 2;
  float* rs = (float*)(ws + off);                 off += (size_t)NTOT * H_ * 4;
  unsigned short* Ah = (unsigned short*)(ws + off); off += (size_t)NTOT * 1536 * 2;
  unsigned short* At = (unsigned short*)(ws + off); off += (size_t)NTOT * 1536 * 2;
  unsigned short* hsp = (unsigned short*)(ws + off); off += (size_t)NTOT * EMB_ * 2;
  unsigned short* tsp = (unsigned short*)(ws + off); off += (size_t)NTOT * EMB_ * 2;  // ~72 MB total

  transpose_bf16_kernel<<<dim3((L_ * H_ + 255) / 256, 1, B_), 256, 0, stream>>>(
      seq, seqT, L_, H_, H_, (long)L_ * H_, (long)H_ * L_);
  transpose_bf16_kernel<<<dim3((1536 * H_ + 255) / 256, 1, 1), 256, 0, stream>>>(
      Wh, WhT, 1536, H_, H_, 0, 0);
  transpose_bf16_kernel<<<dim3((1536 * H_ + 255) / 256, 1, 1), 256, 0, stream>>>(
      Wt, WtT, 1536, H_, H_, 0, 0);
  transpose_bf16_kernel<<<dim3((KBIL * NCLSPAD + 255) / 256, 1, 1), 256, 0, stream>>>(
      Wb, WbT, KBIL, NCLS, NCLSPAD, 0, 0);

  ent_kernel<<<dim3(B_ * E_), 256, 0, stream>>>(seq, att, mpos, ent_emb, ent_att);
  htatt_kernel<<<dim3(B_ * P_), 256, 0, stream>>>(hts, ent_att, htatt);

  // rs: per batch 31 x 24 blocks, each wave does a 32x32 output block (4 WMMA tiles).
  gemm_rs_kernel<<<dim3(P_ / 32, H_ / 32, B_), 32, 0, stream>>>(htatt, seqT, rs);

  build_A_kernel<<<dim3(NTOT), 256, 0, stream>>>(hts, ent_emb, rs, Ah, At);

  // proj: 124 x 24 x 2 blocks of 32x32.
  gemm_proj_kernel<<<dim3(NTOT / 32, EMB_ / 32, 2), 32, 0, stream>>>(
      Ah, At, WhT, WtT, bh, bt, hsp, tsp);

  // logits: 124 blocks, 7 column waves, 2 row tiles per wave, K=49152.
  gemm_logits_kernel<<<dim3(NTOT / 32), 224, 0, stream>>>(hsp, tsp, WbT, bb, out);
}
